// PhaseAttention_29807073034670
// MI455X (gfx1250) — matrix operand order repaired and checked by reading, not compiled, not run on hardware
//
#include <hip/hip_runtime.h>
#include <stdint.h>

// ---------------------------------------------------------------------------
// Types for CDNA5 WMMA
// ---------------------------------------------------------------------------
typedef unsigned short ushort_t;
typedef __attribute__((ext_vector_type(16))) __bf16 v16bf;
typedef __attribute__((ext_vector_type(8)))  float  v8f;

union FragU { v16bf v; uint4 q[2]; };

static __device__ __forceinline__ v8f wmma_bf16(v16bf a, v16bf b, v8f c) {
  // D = A(16x32 bf16) x B(32x16 bf16) + C(16x16 f32)
  return __builtin_amdgcn_wmma_f32_16x16x32_bf16(false, a, false, b, (short)0, c, false, false);
}

static __device__ __forceinline__ ushort_t f2bf(float f) {
  uint32_t u = __float_as_uint(f);
  u += 0x7FFFu + ((u >> 16) & 1u);   // round-to-nearest-even
  return (ushort_t)(u >> 16);
}

// CDNA5 async global->LDS copy (ASYNCcnt-tracked); LDS dest is the low 32 bits
// of the flat shared-aperture address.
static __device__ __forceinline__ void async_g2l_b128(void* lds_ptr, const void* gptr) {
  uint32_t lo = (uint32_t)(uintptr_t)lds_ptr;
  asm volatile("global_load_async_to_lds_b128 %0, %1, off"
               :: "v"(lo), "v"((uint64_t)(uintptr_t)gptr)
               : "memory");
}
static __device__ __forceinline__ void wait_async0() {
  asm volatile("s_wait_asynccnt 0x0" ::: "memory");
}

#define NEGINF (-__builtin_inff())

// ---------------------------------------------------------------------------
// Problem constants
// ---------------------------------------------------------------------------
#define DIMC   1024
#define NH     16
#define HDIM   64
#define NB     2
#define SEQL   4096
#define MTOK   (NB*SEQL)          // 8192 tokens
#define CHUNK  256
#define NCHUNK (SEQL/CHUNK)       // 16
#define VSTR   4608               // padded V^T position stride
#define VPOFF  263                // position pad offset (== 7 mod 8 -> 16B aligned frags)

static constexpr size_t XPLANE = (size_t)MTOK * DIMC;          // 8388608 elems
static constexpr size_t WPLANE = (size_t)DIMC * DIMC;          // 1048576 elems
static constexpr size_t QPLANE = (size_t)NB * NH * SEQL * HDIM;// 8388608 elems
static constexpr size_t VPLANE = (size_t)NB * NH * HDIM * VSTR;// 9437184 elems

static constexpr size_t OFF_XR = 0;
static constexpr size_t OFF_XI = OFF_XR + XPLANE * 2;
static constexpr size_t OFF_W  = OFF_XI + XPLANE * 2;          // 12 planes: {R,I,-I} x {q,k,v,o}
static constexpr size_t OFF_Q  = OFF_W  + 12 * WPLANE * 2;     // qR,qI,kR,kI
static constexpr size_t OFF_VT = OFF_Q  + 4 * QPLANE * 2;      // vtR,vtI (padded)
static constexpr size_t WS_NEED = OFF_VT + 2 * VPLANE * 2;     // ~164 MB

// ---------------------------------------------------------------------------
// Kernel 1: convert z (f32 interleaved complex) -> bf16 planes xr, xi
// ---------------------------------------------------------------------------
__global__ __launch_bounds__(256) void conv_z_kernel(
    const float* __restrict__ z, ushort_t* __restrict__ xr, ushort_t* __restrict__ xi) {
  size_t i = (size_t)blockIdx.x * 256 + threadIdx.x;   // < MTOK*DIMC
  float2 p = ((const float2*)z)[i];
  xr[i] = f2bf(p.x);
  xi[i] = f2bf(p.y);
}

// ---------------------------------------------------------------------------
// Kernel 2: convert+transpose weights: W[k][n][c] f32 -> Wt[n][k] bf16 planes
// For each weight store {Re, Im, -Im} so complex GEMM needs no A/B negation
// (CDNA5 bf16 WMMA only supports C-negate).
// ---------------------------------------------------------------------------
__global__ __launch_bounds__(256) void conv_w_kernel(
    const float* __restrict__ Wq, const float* __restrict__ Wk,
    const float* __restrict__ Wv, const float* __restrict__ Wo,
    ushort_t* __restrict__ Wall) {
  size_t i = (size_t)blockIdx.x * 256 + threadIdx.x;   // < DIMC*DIMC
  int k = (int)(i >> 10), n = (int)(i & 1023);
  size_t t = (size_t)n * DIMC + k;                     // transposed index
  const float* srcs[4] = {Wq, Wk, Wv, Wo};
  #pragma unroll
  for (int w = 0; w < 4; ++w) {
    float2 p = ((const float2*)srcs[w])[i];            // (k*1024+n) complex pair
    ushort_t* base = Wall + (size_t)w * 3 * WPLANE;
    base[t]              = f2bf(p.x);
    base[WPLANE + t]     = f2bf(p.y);
    base[2 * WPLANE + t] = f2bf(-p.y);
  }
}

// ---------------------------------------------------------------------------
// Kernel 3: complex GEMM  Y = X @ W  (bf16 in, f32 accumulate)
//   yr = xr*Wr + xi*(-Wi) ; yi = xr*Wi + xi*Wr
// Workgroup tile 128M x 64N, K-step 32, 8 waves as 4x2, each wave 32x32.
// Double-buffered LDS filled by async global->LDS copies: tile k+1 streams in
// (ASYNCcnt) while tile k feeds 16 WMMAs per wave. Per-wave s_wait_asynccnt 0
// before the barrier publishes each wave's writes to the workgroup.
// mode 0: blockIdx.z selects q/k/v; q/k written [b,h,l,hd] bf16, v written
//         transposed+padded [b,h,hd,pos+VPOFF] bf16.
// mode 1: Wout; writes interleaved f32 complex pairs to d_out.
// ---------------------------------------------------------------------------
__global__ __launch_bounds__(256) void cgemm_kernel(
    const ushort_t* __restrict__ XRp, const ushort_t* __restrict__ XIp,
    const ushort_t* __restrict__ Wall, ushort_t* __restrict__ Qall,
    ushort_t* __restrict__ Vt, float* __restrict__ outF, int mode) {
  __shared__ __align__(16) ushort_t sX[2][2][128 * 40];  // [buf][plane]
  __shared__ __align__(16) ushort_t sW[2][3][64 * 40];   // [buf][plane]

  const int tid = threadIdx.x, lane = tid & 31, wave = tid >> 5;
  const int mw = wave & 3, nw = wave >> 2;
  const int mblk = blockIdx.x, nblk = blockIdx.y;
  const int wsel = (mode == 0) ? (int)blockIdx.z : 3;
  const ushort_t* WR = Wall + (size_t)wsel * 3 * WPLANE;
  const ushort_t* WI = WR + WPLANE;
  const ushort_t* WN = WR + 2 * WPLANE;

  v8f accR[2][2], accI[2][2];
  #pragma unroll
  for (int a = 0; a < 2; ++a)
    #pragma unroll
    for (int c = 0; c < 2; ++c)
      #pragma unroll
      for (int j = 0; j < 8; ++j) { accR[a][c][j] = 0.f; accI[a][c][j] = 0.f; }

  const int lrow = lane & 15, kh = lane >> 4;
  const int xr_row = tid >> 1, xr_half = tid & 1;      // X tile loader mapping
  const int w_row = tid >> 2, w_q = tid & 3;           // W tile loader mapping

  // async-issue one K-tile (32 deep) into LDS buffer `buf`
  auto load_tile = [&](int kb, int buf) {
    const int k0g = kb * 32;
    size_t g = (size_t)(mblk * 128 + xr_row) * DIMC + k0g + xr_half * 16;
    int so = xr_row * 40 + xr_half * 16;
    async_g2l_b128(&sX[buf][0][so],     XRp + g);
    async_g2l_b128(&sX[buf][0][so + 8], XRp + g + 8);
    async_g2l_b128(&sX[buf][1][so],     XIp + g);
    async_g2l_b128(&sX[buf][1][so + 8], XIp + g + 8);
    size_t gw = (size_t)(nblk * 64 + w_row) * DIMC + k0g + w_q * 8;
    int swo = w_row * 40 + w_q * 8;
    async_g2l_b128(&sW[buf][0][swo], WR + gw);
    async_g2l_b128(&sW[buf][1][swo], WI + gw);
    async_g2l_b128(&sW[buf][2][swo], WN + gw);
  };

  load_tile(0, 0);
  wait_async0();
  __syncthreads();

  for (int kb = 0; kb < 32; ++kb) {
    const int cur = kb & 1;
    if (kb < 31) load_tile(kb + 1, cur ^ 1);   // overlaps with WMMAs below

    FragU xa[2][2], wb[3][2];
    #pragma unroll
    for (int mt = 0; mt < 2; ++mt) {               // A frags (interleaved K halves)
      int row = mw * 32 + mt * 16 + lrow;
      int o0 = row * 40 + kh * 8;
      xa[0][mt].q[0] = *(const uint4*)&sX[cur][0][o0];
      xa[0][mt].q[1] = *(const uint4*)&sX[cur][0][o0 + 16];
      xa[1][mt].q[0] = *(const uint4*)&sX[cur][1][o0];
      xa[1][mt].q[1] = *(const uint4*)&sX[cur][1][o0 + 16];
    }
    #pragma unroll
    for (int nt = 0; nt < 2; ++nt) {               // B frags (interleaved K halves)
      int row = nw * 32 + nt * 16 + lrow;
      int o0 = row * 40 + kh * 8;
      #pragma unroll
      for (int p = 0; p < 3; ++p) {
        wb[p][nt].q[0] = *(const uint4*)&sW[cur][p][o0];
        wb[p][nt].q[1] = *(const uint4*)&sW[cur][p][o0 + 16];
      }
    }
    #pragma unroll
    for (int mt = 0; mt < 2; ++mt)
      #pragma unroll
      for (int nt = 0; nt < 2; ++nt) {
        accR[mt][nt] = wmma_bf16(xa[0][mt].v, wb[0][nt].v, accR[mt][nt]); // xr*Wr
        accR[mt][nt] = wmma_bf16(xa[1][mt].v, wb[2][nt].v, accR[mt][nt]); // xi*(-Wi)
        accI[mt][nt] = wmma_bf16(xa[0][mt].v, wb[1][nt].v, accI[mt][nt]); // xr*Wi
        accI[mt][nt] = wmma_bf16(xa[1][mt].v, wb[0][nt].v, accI[mt][nt]); // xi*Wr
      }

    wait_async0();     // my next-tile writes have landed
    __syncthreads();   // everyone's writes visible; everyone done reading cur
  }

  // Epilogue. D-frag element (m,n): lane = n + 16*(m>=8), vgpr r = m&7.
  const int n_l = lane & 15, hi = lane >> 4;
  #pragma unroll
  for (int mt = 0; mt < 2; ++mt)
    #pragma unroll
    for (int nt = 0; nt < 2; ++nt)
      #pragma unroll
      for (int r = 0; r < 8; ++r) {
        int gm = mblk * 128 + mw * 32 + mt * 16 + r + 8 * hi;
        int gn = nblk * 64 + nw * 32 + nt * 16 + n_l;
        float yr = accR[mt][nt][r], yi = accI[mt][nt][r];
        if (mode == 1) {
          *(float2*)(outF + ((size_t)gm * DIMC + gn) * 2) = make_float2(yr, yi);
        } else {
          int bb = gm >> 12, ll = gm & 4095, hh = gn >> 6, hd = gn & 63;
          if (wsel < 2) {  // q or k: [b,h,l,hd]
            size_t idx = (((size_t)bb * NH + hh) * SEQL + ll) * HDIM + hd;
            ushort_t* oRp = Qall + (size_t)wsel * 2 * QPLANE;
            oRp[idx]          = f2bf(yr);
            oRp[QPLANE + idx] = f2bf(yi);
          } else {         // v: transposed + padded [b,h,hd,pos]
            size_t idx = (((size_t)bb * NH + hh) * HDIM + hd) * VSTR + (ll + VPOFF);
            Vt[idx]          = f2bf(yr);
            Vt[VPLANE + idx] = f2bf(yi);
          }
        }
      }
}

// ---------------------------------------------------------------------------
// Kernel 4: windowed flash attention (online softmax), all WMMA.
// grid = (NCHUNK, NH, NB); block = 256 (8 waves); wave handles 16-query tiles.
// ---------------------------------------------------------------------------
__global__ __launch_bounds__(256) void attn_kernel(
    const ushort_t* __restrict__ qR, const ushort_t* __restrict__ qI,
    const ushort_t* __restrict__ kR, const ushort_t* __restrict__ kI,
    const ushort_t* __restrict__ vR, const ushort_t* __restrict__ vI,
    ushort_t* __restrict__ aR, ushort_t* __restrict__ aI) {
  __shared__ __align__(16) ushort_t P[8][16 * 32];   // per-wave P tile (bf16)

  const int tid = threadIdx.x, lane = tid & 31, wave = tid >> 5;
  const int lrow = lane & 15, kh = lane >> 4;
  const int cblk = blockIdx.x, h = blockIdx.y, b = blockIdx.z;
  const int cs = cblk * CHUNK;
  const size_t bh = (size_t)b * NH + h;
  const ushort_t* qRb = qR + bh * (SEQL * HDIM);
  const ushort_t* qIb = qI + bh * (SEQL * HDIM);
  const ushort_t* kRb = kR + bh * (SEQL * HDIM);
  const ushort_t* kIb = kI + bh * (SEQL * HDIM);
  const ushort_t* vRb = vR + bh * (HDIM * VSTR);
  const ushort_t* vIb = vI + bh * (HDIM * VSTR);
  ushort_t* Pw = P[wave];

  for (int rt = wave; rt < CHUNK / 16; rt += 8) {
    // Q A-fragments: 16 queries x 64 hd, two 32-K chunks, r/i
    FragU qa[2][2];
    {
      size_t qoff = (size_t)(cs + rt * 16 + lrow) * HDIM;
      #pragma unroll
      for (int hhalf = 0; hhalf < 2; ++hhalf) {
        int o = hhalf * 32 + kh * 8;
        qa[0][hhalf].q[0] = *(const uint4*)(qRb + qoff + o);
        qa[0][hhalf].q[1] = *(const uint4*)(qRb + qoff + o + 16);
        qa[1][hhalf].q[0] = *(const uint4*)(qIb + qoff + o);
        qa[1][hhalf].q[1] = *(const uint4*)(qIb + qoff + o + 16);
      }
    }

    v8f oRacc[4], oIacc[4];
    float mrun[8], lrun[8];
    #pragma unroll
    for (int d = 0; d < 4; ++d)
      #pragma unroll
      for (int j = 0; j < 8; ++j) { oRacc[d][j] = 0.f; oIacc[d][j] = 0.f; }
    #pragma unroll
    for (int r = 0; r < 8; ++r) { mrun[r] = NEGINF; lrun[r] = 0.f; }

    // only key blocks overlapping [qrow, qrow+255] for qrow in this 16-row tile
    int bi0 = rt >> 1;
    int bi1 = (rt * 16 + 270) >> 5; if (bi1 > 15) bi1 = 15;

    for (int blk = bi0; blk <= bi1; ++blk) {
      // ---- scores: two 16x16 tiles over this 32-key block ----
      v8f s[2];
      #pragma unroll
      for (int t = 0; t < 2; ++t) {
        int key = blk * 32 + t * 16 + lrow;      // window key index for this lane
        int kp = cs - 255 + key;                 // absolute position
        int kcl = kp < 0 ? 0 : (kp > SEQL - 1 ? SEQL - 1 : kp);
        size_t koff = (size_t)kcl * HDIM;
        v8f acc;
        #pragma unroll
        for (int j = 0; j < 8; ++j) acc[j] = 0.f;
        #pragma unroll
        for (int hhalf = 0; hhalf < 2; ++hhalf) {
          int o = hhalf * 32 + kh * 8;           // B layout: interleaved K halves
          FragU krf, kif;
          krf.q[0] = *(const uint4*)(kRb + koff + o);
          krf.q[1] = *(const uint4*)(kRb + koff + o + 16);
          kif.q[0] = *(const uint4*)(kIb + koff + o);
          kif.q[1] = *(const uint4*)(kIb + koff + o + 16);
          acc = wmma_bf16(qa[0][hhalf].v, krf.v, acc);   // qr . kr
          acc = wmma_bf16(qa[1][hhalf].v, kif.v, acc);   // qi . ki
        }
        s[t] = acc;
      }

      // ---- mask + online softmax (row stats live in 16-lane halves) ----
      float alpha[8];
      #pragma unroll
      for (int r = 0; r < 8; ++r) {
        int qrow = rt * 16 + r + 8 * kh;
        int ki0 = blk * 32 + lrow, ki1 = ki0 + 16;
        float s0 = s[0][r] * 0.125f, s1 = s[1][r] * 0.125f;  // HD^-0.5
        bool v0 = (ki0 >= qrow) && (ki0 <= qrow + 255) && (cs - 255 + ki0 >= 0);
        bool v1 = (ki1 >= qrow) && (ki1 <= qrow + 255) && (cs - 255 + ki1 >= 0);
        s0 = v0 ? s0 : NEGINF;
        s1 = v1 ? s1 : NEGINF;
        float tm = fmaxf(s0, s1);
        tm = fmaxf(tm, __shfl_xor(tm, 1, 16));
        tm = fmaxf(tm, __shfl_xor(tm, 2, 16));
        tm = fmaxf(tm, __shfl_xor(tm, 4, 16));
        tm = fmaxf(tm, __shfl_xor(tm, 8, 16));
        float mnew = fmaxf(mrun[r], tm);
        float msafe = (mnew == NEGINF) ? 0.f : mnew;
        float a = __expf(mrun[r] - msafe);       // 0 when mrun = -inf
        mrun[r] = mnew;
        float p0 = __expf(s0 - msafe), p1 = __expf(s1 - msafe);
        float rs = p0 + p1;
        rs += __shfl_xor(rs, 1, 16);
        rs += __shfl_xor(rs, 2, 16);
        rs += __shfl_xor(rs, 4, 16);
        rs += __shfl_xor(rs, 8, 16);
        lrun[r] = lrun[r] * a + rs;
        alpha[r] = a;
        Pw[(r + 8 * kh) * 32 + lrow]      = f2bf(p0);   // D-layout -> LDS row-major
        Pw[(r + 8 * kh) * 32 + 16 + lrow] = f2bf(p1);
      }

      #pragma unroll
      for (int d = 0; d < 4; ++d)
        #pragma unroll
        for (int r = 0; r < 8; ++r) { oRacc[d][r] *= alpha[r]; oIacc[d][r] *= alpha[r]; }

      asm volatile("s_wait_dscnt 0x0" ::: "memory");   // wave-local LDS RAW fence

      FragU pa;                                 // P as A-frag (16x32 over keys)
      {
        int o = lrow * 32 + kh * 16;
        pa.q[0] = *(const uint4*)&Pw[o];
        pa.q[1] = *(const uint4*)&Pw[o + 8];
      }

      // ---- P . V via transposed padded V planes ----
      #pragma unroll
      for (int d = 0; d < 4; ++d) {
        size_t vo = (size_t)(d * 16 + lrow) * VSTR
                  + (size_t)(cs + (VPOFF - 255) + blk * 32 + kh * 16);
        FragU vrf, vif;
        vrf.q[0] = *(const uint4*)(vRb + vo);
        vrf.q[1] = *(const uint4*)(vRb + vo + 8);
        vif.q[0] = *(const uint4*)(vIb + vo);
        vif.q[1] = *(const uint4*)(vIb + vo + 8);
        oRacc[d] = wmma_bf16(pa.v, vrf.v, oRacc[d]);
        oIacc[d] = wmma_bf16(pa.v, vif.v, oIacc[d]);
      }
    }

    // ---- normalize and store as bf16 planes [token][h*64+d] ----
    #pragma unroll
    for (int r = 0; r < 8; ++r) lrun[r] = 1.f / lrun[r];
    #pragma unroll
    for (int d = 0; d < 4; ++d)
      #pragma unroll
      for (int r = 0; r < 8; ++r) {
        size_t token = (size_t)b * SEQL + cs + rt * 16 + r + 8 * kh;
        int col = h * HDIM + d * 16 + lrow;
        size_t idx = token * DIMC + col;
        aR[idx] = f2bf(oRacc[d][r] * lrun[r]);
        aI[idx] = f2bf(oIacc[d][r] * lrun[r]);
      }
  }
}

// ---------------------------------------------------------------------------
// Host-side launch
// ---------------------------------------------------------------------------
extern "C" void kernel_launch(void* const* d_in, const int* in_sizes, int n_in,
                              void* d_out, int out_size, void* d_ws, size_t ws_size,
                              hipStream_t stream) {
  (void)in_sizes; (void)n_in; (void)out_size;
  if (ws_size < WS_NEED) return;   // workspace too small; cannot proceed safely

  const float* z  = (const float*)d_in[0];
  const float* Wq = (const float*)d_in[1];
  const float* Wk = (const float*)d_in[2];
  const float* Wv = (const float*)d_in[3];
  const float* Wo = (const float*)d_in[4];

  char* ws = (char*)d_ws;
  ushort_t* XR   = (ushort_t*)(ws + OFF_XR);
  ushort_t* XI   = (ushort_t*)(ws + OFF_XI);
  ushort_t* Wall = (ushort_t*)(ws + OFF_W);
  ushort_t* Qall = (ushort_t*)(ws + OFF_Q);
  ushort_t* Vt   = (ushort_t*)(ws + OFF_VT);

  // zero padded V^T planes so out-of-window fragment loads are clean
  hipMemsetAsync(Vt, 0, 2 * VPLANE * sizeof(ushort_t), stream);

  conv_z_kernel<<<dim3((unsigned)(XPLANE / 256)), 256, 0, stream>>>(z, XR, XI);
  conv_w_kernel<<<dim3((unsigned)(WPLANE / 256)), 256, 0, stream>>>(Wq, Wk, Wv, Wo, Wall);

  // q,k,v projections (3 weights in blockIdx.z)
  cgemm_kernel<<<dim3(MTOK / 128, DIMC / 64, 3), 256, 0, stream>>>(
      XR, XI, Wall, Qall, Vt, nullptr, 0);

  // windowed attention -> bf16 planes into XR/XI region (reuse)
  attn_kernel<<<dim3(NCHUNK, NH, NB), 256, 0, stream>>>(
      Qall, Qall + QPLANE, Qall + 2 * QPLANE, Qall + 3 * QPLANE,
      Vt, Vt + VPLANE, XR, XI);

  // output projection -> interleaved f32 complex to d_out
  cgemm_kernel<<<dim3(MTOK / 128, DIMC / 64, 1), 256, 0, stream>>>(
      XR, XI, Wall, Qall, Vt, (float*)d_out, 1);
}
